// NeuromorphicLM_74929999446024
// MI455X (gfx1250) — compile-verified, matrix operand order checked by reading, not run on hardware
//
#include <hip/hip_runtime.h>
#include <hip/hip_bf16.h>

static constexpr int kBS    = 4;
static constexpr int kSEQ   = 512;
static constexpr int kC     = 8;
static constexpr int kDC    = 64;
static constexpr int kD     = 512;
static constexpr int kB     = 8;
static constexpr int kM     = 64;
static constexpr int kL     = 2;
static constexpr int kE     = 128;
static constexpr int kV     = 32000;
static constexpr int kNT    = kBS * kSEQ;          // 2048 tokens
static constexpr float kETA = 0.1f;                // ETA_PM
static constexpr float kPCMW= 0.1f;                // pcm_pred_weight

typedef __bf16 bf16_t;
typedef __attribute__((ext_vector_type(8)))  __bf16 v8bf;
typedef __attribute__((ext_vector_type(16))) __bf16 v16bf;
typedef __attribute__((ext_vector_type(8)))  float  v8f;

// ---- gfx1250 async global->LDS copy support (guarded; falls back cleanly) ----
#if defined(__gfx1250__) && __has_builtin(__builtin_amdgcn_global_load_async_to_lds_b128)
#define GEMM_ASYNC_LDS 1
typedef int v4i_t __attribute__((vector_size(4 * sizeof(int))));
#if __has_builtin(__builtin_amdgcn_s_wait_asynccnt)
#define WAIT_ASYNC(n) __builtin_amdgcn_s_wait_asynccnt(n)
#else
#define WAIT_ASYNC(n) asm volatile("s_wait_asynccnt %0" :: "n"(n) : "memory")
#endif
#endif

__device__ __forceinline__ bf16_t f2bf(float f) {
  unsigned u = __builtin_bit_cast(unsigned, f);
  unsigned r = (u + 0x7FFFu + ((u >> 16) & 1u)) >> 16;
  unsigned short s = (unsigned short)r;
  return __builtin_bit_cast(bf16_t, s);
}
__device__ __forceinline__ float sigm(float x){ return 1.0f / (1.0f + __expf(-x)); }
__device__ __forceinline__ float gelu(float x){
  float x3 = x * x * x;
  return 0.5f * x * (1.0f + tanhf(0.7978845608028654f * (x + 0.044715f * x3)));
}

// ---------------- embedding gather + positional ----------------
__global__ void k_zero_aux(float* aux){ if (threadIdx.x == 0 && blockIdx.x == 0) *aux = 0.0f; }

__global__ void k_embed(const int* __restrict__ ids, const float* __restrict__ emb,
                        const float* __restrict__ pos, float* __restrict__ X0,
                        float* __restrict__ H){
  int t = blockIdx.x; int n = t % kSEQ;
  int id = ids[t];
  for (int d = threadIdx.x; d < kD; d += blockDim.x){
    float v = emb[(long)id * kD + d] + pos[(long)n * kD + d];
    X0[(long)t * kD + d] = v;
    H [(long)t * kD + d] = v;
  }
}

// ---------------- scan layer: LN + v/g projections ----------------
__global__ void k_scan_pre(const float* __restrict__ X, const float* __restrict__ Wv,
                           const float* __restrict__ Wg, const float* __restrict__ lng,
                           const float* __restrict__ lnb, float* __restrict__ Vb,
                           float* __restrict__ Gb){
  __shared__ float h[kDC];
  __shared__ float red[kDC];
  __shared__ float stat[2];
  int idx = blockIdx.x;
  int t = idx / kC, c = idx % kC;
  int tid = threadIdx.x;
  long base = (long)t * kD + c * kDC;
  float x = X[base + tid];
  red[tid] = x; __syncthreads();
  for (int s = 32; s > 0; s >>= 1){ if (tid < s) red[tid] += red[tid + s]; __syncthreads(); }
  if (tid == 0) stat[0] = red[0] * (1.0f / kDC);
  __syncthreads();
  float mu = stat[0];
  float dxm = x - mu;
  red[tid] = dxm * dxm; __syncthreads();
  for (int s = 32; s > 0; s >>= 1){ if (tid < s) red[tid] += red[tid + s]; __syncthreads(); }
  if (tid == 0) stat[1] = red[0] * (1.0f / kDC);
  __syncthreads();
  float hv = dxm * rsqrtf(stat[1] + 1e-5f) * lng[tid] + lnb[tid];
  h[tid] = hv; __syncthreads();
  const float* wv = Wv + (long)c * kDC * kDC;
  const float* wg = Wg + (long)c * kDC * kDC;
  float av = 0.0f, ag = 0.0f;
  for (int d = 0; d < kDC; ++d){
    float hd = h[d];
    av += hd * wv[d * kDC + tid];
    ag += hd * wg[d * kDC + tid];
  }
  Vb[base + tid] = av;
  Gb[base + tid] = sigm(ag);
}

// ---------------- scan layer: sequential recurrence over N ----------------
__global__ void k_scan_seq(const float* __restrict__ Vb, const float* __restrict__ Gb,
                           const float* __restrict__ alog, float* __restrict__ Ub){
  int bc = blockIdx.x; int b = bc / kC, c = bc % kC;
  int e = threadIdx.x;
  float a = sigm(alog[c * kDC + e]);
  float s = 0.0f;
  long base = ((long)b * kSEQ) * kD + c * kDC + e;
  for (int n = 0; n < kSEQ; ++n){
    long i = base + (long)n * kD;
    s = a * s + Vb[i];
    Ub[i] = Gb[i] * s;
  }
}

// ---------------- scan layer: per-column MLP, residual add ----------------
__global__ void k_scan_mlp(const float* __restrict__ Ub, const float* __restrict__ W1,
                           const float* __restrict__ W2, float* __restrict__ X){
  __shared__ float u[kDC];
  __shared__ float tt[kE];
  int idx = blockIdx.x; int t = idx / kC, c = idx % kC;
  int tid = threadIdx.x;
  long base = (long)t * kD + c * kDC;
  if (tid < kDC) u[tid] = Ub[base + tid];
  __syncthreads();
  const float* w1 = W1 + (long)c * kDC * kE;
  float z = 0.0f;
  for (int d = 0; d < kDC; ++d) z += u[d] * w1[d * kE + tid];
  tt[tid] = gelu(z);
  __syncthreads();
  if (tid < kDC){
    const float* w2 = W2 + (long)c * kE * kDC;
    float y = 0.0f;
    for (int e = 0; e < kE; ++e) y += tt[e] * w2[e * kDC + tid];
    X[base + tid] += y;
  }
}

// ---------------- PCM: predict, surprise, modulate, aux loss ----------------
__global__ void k_pcm(float* __restrict__ H, const float* __restrict__ X0,
                      const float* __restrict__ Wpred, const float* __restrict__ bpred,
                      const float* __restrict__ Wgain, float* __restrict__ Surp,
                      float* __restrict__ aux){
  __shared__ float hh[kDC], ss[kDC], red[kDC];
  int idx = blockIdx.x; int t = idx / kC, c = idx % kC;
  int tid = threadIdx.x;
  long base = (long)t * kD + c * kDC;
  hh[tid] = H[base + tid];
  __syncthreads();
  const float* wp = Wpred + (long)c * kDC * kDC;
  float z = bpred[c * kDC + tid];
  for (int d = 0; d < kDC; ++d) z += hh[d] * wp[d * kDC + tid];
  float s = X0[base + tid] - z;
  ss[tid] = s;
  red[tid] = s * s;
  __syncthreads();
  for (int r = 32; r > 0; r >>= 1){ if (tid < r) red[tid] += red[tid + r]; __syncthreads(); }
  if (tid == 0) atomicAdd(aux, kPCMW * red[0] / (float)((long)kNT * kD));
  const float* wg = Wgain + (long)c * kDC * kDC;
  float g = 0.0f;
  for (int d = 0; d < kDC; ++d) g += ss[d] * wg[d * kDC + tid];
  H[base + tid] = hh[tid] * (1.0f + tanhf(g));
  Surp[base + tid] = s;
}

// ---------------- seed / write-candidate heads ----------------
__global__ void k_seedw(const float* __restrict__ H, const float* __restrict__ W,
                        const float* __restrict__ bvec, float* __restrict__ Seed,
                        float* __restrict__ Wc){
  __shared__ float hh[kDC];
  int idx = blockIdx.x; int t = idx / kC, c = idx % kC;
  int tid = threadIdx.x;
  long base = (long)t * kD + c * kDC;
  if (tid < kDC) hh[tid] = H[base + tid];
  __syncthreads();
  const float* w = W + (long)c * kDC * (2 * kDC);
  float v = bvec[c * 2 * kDC + tid];
  for (int d = 0; d < kDC; ++d) v += hh[d] * w[d * 2 * kDC + tid];
  if (tid < kDC) Seed[base + tid] = v;
  else           Wc[base + (tid - kDC)] = v;
}

// ---------------- dtype conversions ----------------
__global__ void k_tanh_bf(const float* __restrict__ in, bf16_t* __restrict__ out, long n){
  for (long i = blockIdx.x * (long)blockDim.x + threadIdx.x; i < n;
       i += (long)gridDim.x * blockDim.x)
    out[i] = f2bf(tanhf(in[i]));
}
__global__ void k_f32_bf(const float* __restrict__ in, bf16_t* __restrict__ out, long n){
  for (long i = blockIdx.x * (long)blockDim.x + threadIdx.x; i < n;
       i += (long)gridDim.x * blockDim.x)
    out[i] = f2bf(in[i]);
}
__global__ void k_wwrT(const float* __restrict__ in, bf16_t* __restrict__ out){
  long n = (long)kB * kD * kD;
  for (long i = blockIdx.x * (long)blockDim.x + threadIdx.x; i < n;
       i += (long)gridDim.x * blockDim.x){
    long k = i / ((long)kD * kD);
    long r = i - k * kD * kD;
    long d = r / kD, f = r - d * kD;
    out[k * kD * kD + f * kD + d] = f2bf(in[i]);   // transpose: [k][f][d]
  }
}

// ---------------- NT bf16 WMMA GEMM: C[M,N] = alpha * A[M,K] . B[N,K]^T ----------------
// Block: 256 threads = 8 waves, block tile 32(M) x 256(N); wave tile 16 x 64 (4 accumulators).
// Fragment layout per CDNA5 ISA 7.12.2: lanes 0-15 hold K {0..7,16..23}, lanes 16-31 K {8..15,24..31}.
// When available, A/B tiles are staged through LDS with double-buffered
// GLOBAL_LOAD_ASYNC_TO_LDS_B128 (ASYNCcnt-tracked DMA; 5 async instrs per wave per
// 32-wide K chunk -> s_wait_asynccnt 5 overlaps next chunk's copy with WMMA).
__global__ void __launch_bounds__(256)
k_gemm_nt(const bf16_t* __restrict__ A, const bf16_t* __restrict__ Bm,
          float* __restrict__ Cm, int K, int ldc, long strideB, long strideC, float alpha){
  int lane = threadIdx.x & 31;
  int wave = threadIdx.x >> 5;
  int wm = wave >> 2;                  // 0..1
  int wn = wave & 3;                   // 0..3
  int c0 = (lane >= 16) ? 8 : 0;
  v8f acc0 = {}, acc1 = {}, acc2 = {}, acc3 = {};
  float* Cp = Cm + (long)blockIdx.z * strideC;
#ifdef GEMM_ASYNC_LDS
  __shared__ bf16_t Ab[2][32 * 32];
  __shared__ bf16_t Bb[2][256 * 32];
  int tid = threadIdx.x;
  int mBase = blockIdx.x * 32;
  int nBase = blockIdx.y * 256;
  const bf16_t* Ap = A  + (long)mBase * K;
  const bf16_t* Bp = Bm + (long)blockIdx.z * strideB + (long)nBase * K;
  int nk = K >> 5;
  auto issue = [&](int step, int pb){
    int kk = step << 5;
    // B tile: 256 rows x 32 cols, 4 b128 async loads per lane (4 instr / wave)
    #pragma unroll
    for (int i = 0; i < 4; ++i){
      int j = i * 256 + tid;
      int r = j >> 2, q = j & 3;
      const bf16_t* g = Bp + (long)r * K + kk + q * 8;
      bf16_t* l = &Bb[pb][r * 32 + q * 8];
      __builtin_amdgcn_global_load_async_to_lds_b128(
          (v4i_t*)(void*)g, (v4i_t*)(void*)l, 0, 0);
    }
    // A tile: 32 rows x 32 cols, lanes 0-15 of every wave (1 instr / wave)
    if (lane < 16){
      int j = wave * 16 + lane;
      int r = j >> 2, q = j & 3;
      const bf16_t* g = Ap + (long)r * K + kk + q * 8;
      bf16_t* l = &Ab[pb][r * 32 + q * 8];
      __builtin_amdgcn_global_load_async_to_lds_b128(
          (v4i_t*)(void*)g, (v4i_t*)(void*)l, 0, 0);
    }
  };
  issue(0, 0);
  for (int step = 0; step < nk; ++step){
    int pb = step & 1;
    if (step + 1 < nk){ issue(step + 1, pb ^ 1); WAIT_ASYNC(5); }
    else              { WAIT_ASYNC(0); }
    __syncthreads();                       // all waves' chunk data visible in LDS
    const bf16_t* ar = &Ab[pb][(wm * 16 + (lane & 15)) * 32 + c0];
    v8bf alo = *(const v8bf*)ar;
    v8bf ahi = *(const v8bf*)(ar + 16);
    v16bf af = __builtin_shufflevector(alo, ahi, 0,1,2,3,4,5,6,7,8,9,10,11,12,13,14,15);
    const bf16_t* b0 = &Bb[pb][(wn * 64 +  0 + (lane & 15)) * 32 + c0];
    const bf16_t* b1 = &Bb[pb][(wn * 64 + 16 + (lane & 15)) * 32 + c0];
    const bf16_t* b2 = &Bb[pb][(wn * 64 + 32 + (lane & 15)) * 32 + c0];
    const bf16_t* b3 = &Bb[pb][(wn * 64 + 48 + (lane & 15)) * 32 + c0];
    v8bf l0 = *(const v8bf*)b0, h0 = *(const v8bf*)(b0 + 16);
    v8bf l1 = *(const v8bf*)b1, h1 = *(const v8bf*)(b1 + 16);
    v8bf l2 = *(const v8bf*)b2, h2 = *(const v8bf*)(b2 + 16);
    v8bf l3 = *(const v8bf*)b3, h3 = *(const v8bf*)(b3 + 16);
    v16bf bf0 = __builtin_shufflevector(l0, h0, 0,1,2,3,4,5,6,7,8,9,10,11,12,13,14,15);
    v16bf bf1 = __builtin_shufflevector(l1, h1, 0,1,2,3,4,5,6,7,8,9,10,11,12,13,14,15);
    v16bf bf2 = __builtin_shufflevector(l2, h2, 0,1,2,3,4,5,6,7,8,9,10,11,12,13,14,15);
    v16bf bf3 = __builtin_shufflevector(l3, h3, 0,1,2,3,4,5,6,7,8,9,10,11,12,13,14,15);
    acc0 = __builtin_amdgcn_wmma_f32_16x16x32_bf16(false, af, false, bf0, (short)0, acc0, false, false);
    acc1 = __builtin_amdgcn_wmma_f32_16x16x32_bf16(false, af, false, bf1, (short)0, acc1, false, false);
    acc2 = __builtin_amdgcn_wmma_f32_16x16x32_bf16(false, af, false, bf2, (short)0, acc2, false, false);
    acc3 = __builtin_amdgcn_wmma_f32_16x16x32_bf16(false, af, false, bf3, (short)0, acc3, false, false);
    __syncthreads();                       // done reading this buffer
  }
#else
  int rowA  = blockIdx.x * 32 + wm * 16 + (lane & 15);
  int colB0 = blockIdx.y * 256 + wn * 64 + (lane & 15);
  const bf16_t* Bp = Bm + (long)blockIdx.z * strideB;
  const bf16_t* aRow  = A  + (long)rowA * K;
  const bf16_t* bRow0 = Bp + (long)(colB0     ) * K;
  const bf16_t* bRow1 = Bp + (long)(colB0 + 16) * K;
  const bf16_t* bRow2 = Bp + (long)(colB0 + 32) * K;
  const bf16_t* bRow3 = Bp + (long)(colB0 + 48) * K;
  for (int kk = 0; kk < K; kk += 32){
    __builtin_prefetch(aRow + kk + c0 + 64, 0, 3);     // global_prefetch_b8
    v8bf alo = *(const v8bf*)(aRow + kk + c0);
    v8bf ahi = *(const v8bf*)(aRow + kk + c0 + 16);
    v16bf af = __builtin_shufflevector(alo, ahi, 0,1,2,3,4,5,6,7,8,9,10,11,12,13,14,15);
    v8bf b0l = *(const v8bf*)(bRow0 + kk + c0);
    v8bf b0h = *(const v8bf*)(bRow0 + kk + c0 + 16);
    v16bf bf0 = __builtin_shufflevector(b0l, b0h, 0,1,2,3,4,5,6,7,8,9,10,11,12,13,14,15);
    v8bf b1l = *(const v8bf*)(bRow1 + kk + c0);
    v8bf b1h = *(const v8bf*)(bRow1 + kk + c0 + 16);
    v16bf bf1 = __builtin_shufflevector(b1l, b1h, 0,1,2,3,4,5,6,7,8,9,10,11,12,13,14,15);
    v8bf b2l = *(const v8bf*)(bRow2 + kk + c0);
    v8bf b2h = *(const v8bf*)(bRow2 + kk + c0 + 16);
    v16bf bf2 = __builtin_shufflevector(b2l, b2h, 0,1,2,3,4,5,6,7,8,9,10,11,12,13,14,15);
    v8bf b3l = *(const v8bf*)(bRow3 + kk + c0);
    v8bf b3h = *(const v8bf*)(bRow3 + kk + c0 + 16);
    v16bf bf3 = __builtin_shufflevector(b3l, b3h, 0,1,2,3,4,5,6,7,8,9,10,11,12,13,14,15);
    acc0 = __builtin_amdgcn_wmma_f32_16x16x32_bf16(false, af, false, bf0, (short)0, acc0, false, false);
    acc1 = __builtin_amdgcn_wmma_f32_16x16x32_bf16(false, af, false, bf1, (short)0, acc1, false, false);
    acc2 = __builtin_amdgcn_wmma_f32_16x16x32_bf16(false, af, false, bf2, (short)0, acc2, false, false);
    acc3 = __builtin_amdgcn_wmma_f32_16x16x32_bf16(false, af, false, bf3, (short)0, acc3, false, false);
  }
#endif
  // C/D layout: lane -> N = lane&15, M = (lane>>4)*8 + vgpr index
  int nIdx  = lane & 15;
  int mBase2 = (lane >> 4) * 8;
  int cRow = blockIdx.x * 32 + wm * 16 + mBase2;
  int cCol = blockIdx.y * 256 + wn * 64 + nIdx;
  for (int v = 0; v < 8; ++v){
    long r = (long)(cRow + v) * ldc;
    Cp[r + cCol     ] = alpha * acc0[v];
    Cp[r + cCol + 16] = alpha * acc1[v];
    Cp[r + cCol + 32] = alpha * acc2[v];
    Cp[r + cCol + 48] = alpha * acc3[v];
  }
}

// ---------------- causal prefix sum over n, per (b,bank) ----------------
__global__ void k_cumsum(float* __restrict__ buf){
  int b = blockIdx.x / kB, k = blockIdx.x % kB;
  int f = threadIdx.x;
  float acc = 0.0f;
  long base = (((long)b * kSEQ) * kB + k) * kD + f;
  for (int n = 0; n < kSEQ; ++n){
    long i = base + (long)n * kB * kD;
    acc += buf[i];
    buf[i] = acc;
  }
}

// ---------------- slot inverse norms ----------------
__global__ void k_slotnorm(const float* __restrict__ slots, float* __restrict__ InvN){
  __shared__ float red[64];
  int s = blockIdx.x;
  int tid = threadIdx.x;
  const float* p = slots + (long)s * kD;
  float acc = 0.0f;
  for (int d = tid; d < kD; d += 64) acc += p[d] * p[d];
  red[tid] = acc; __syncthreads();
  for (int r = 32; r > 0; r >>= 1){ if (tid < r) red[tid] += red[tid + r]; __syncthreads(); }
  if (tid == 0) InvN[s] = 1.0f / (sqrtf(red[0]) + 1e-6f);
}

// ---------------- EM novelty-gated write deltas ----------------
__global__ void __launch_bounds__(256)
k_em_write(const float* __restrict__ H, const float* __restrict__ Wc,
           const float* __restrict__ wnovW, const float* __restrict__ wnovB,
           const float* __restrict__ slots, const float* __restrict__ InvN,
           float* __restrict__ DeltaEM){
  __shared__ float Hs[kD], Ws[kD];
  __shared__ float invwc_s;
  int t = blockIdx.x; int b = t / kSEQ;
  int tid = threadIdx.x;
  int lane = tid & 31, k = tid >> 5;       // one wave per bank
  for (int d = tid; d < kD; d += 256){ Hs[d] = H[(long)t*kD + d]; Ws[d] = Wc[(long)t*kD + d]; }
  __syncthreads();
  float sq = 0.0f;
  for (int d = lane; d < kD; d += 32) sq += Ws[d] * Ws[d];
  for (int m = 16; m > 0; m >>= 1) sq += __shfl_xor(sq, m, 32);
  if (tid == 0) invwc_s = 1.0f / (sqrtf(sq) + 1e-6f);
  float p = 0.0f;
  for (int d = lane; d < kD; d += 32) p += Hs[d] * wnovW[(long)d * kB + k];
  for (int m = 16; m > 0; m >>= 1) p += __shfl_xor(p, m, 32);
  float wnov = sigm(p + wnovB[k]);
  __syncthreads();
  float invwc = invwc_s;
  float mx = -1e30f;
  for (int m0 = lane; m0 < kM; m0 += 32){
    const float* sp = slots + (((long)b * kB + k) * kM + m0) * kD;
    float dp = 0.0f;
    for (int d = 0; d < kD; ++d) dp += Ws[d] * sp[d];
    float sim = dp * invwc * InvN[((long)b * kB + k) * kM + m0];
    mx = fmaxf(mx, sim);
  }
  for (int m = 16; m > 0; m >>= 1) mx = fmaxf(mx, __shfl_xor(mx, m, 32));
  float novel = wnov * fminf(fmaxf(1.0f - mx, 0.0f), 1.0f);
  for (int d = lane; d < kD; d += 32)
    DeltaEM[(((long)t) * kB + k) * kD + d] = novel * Ws[d];
}

// ---------------- PM gated read + integrate (H += pm + cum_em) ----------------
__global__ void __launch_bounds__(256)
k_pm_read(float* __restrict__ H, const float* __restrict__ Wkey,
          const float* __restrict__ pmState, const float* __restrict__ CumPM,
          const float* __restrict__ CumEM){
  __shared__ float Hs[kD];
  __shared__ float gate[kB];
  int t = blockIdx.x; int b = t / kSEQ;
  int tid = threadIdx.x; int lane = tid & 31, k = tid >> 5;
  for (int d = tid; d < kD; d += 256) Hs[d] = H[(long)t * kD + d];
  __syncthreads();
  float p = 0.0f;
  for (int d = lane; d < kD; d += 32) p += Hs[d] * Wkey[(long)d * kB + k];
  for (int m = 16; m > 0; m >>= 1) p += __shfl_xor(p, m, 32);
  if (lane == 0) gate[k] = sigm(p);
  __syncthreads();
  for (int d = tid; d < kD; d += 256){
    float acc = Hs[d];
    for (int kk = 0; kk < kB; ++kk){
      long i = (((long)t) * kB + kk) * kD + d;
      acc += gate[kk] * tanhf(pmState[(((long)b) * kB + kk) * kD + d] + CumPM[i]) + CumEM[i];
    }
    H[(long)t * kD + d] = acc;
  }
}

// ---------------- EM trail read (2-step attention per bank), H += sum_k r ----------------
__global__ void __launch_bounds__(256)
k_em_read(const float* __restrict__ Seed, const float* __restrict__ slots,
          float* __restrict__ H){
  __shared__ float r[kD], accv[kD], att[kM], red[256];
  int t = blockIdx.x; int b = t / kSEQ;
  int tid = threadIdx.x;
  const float scale = 0.044194173824159216f;   // 1/sqrt(512)
  for (int d = tid; d < kD; d += 256) accv[d] = 0.0f;
  for (int k = 0; k < kB; ++k){
    for (int d = tid; d < kD; d += 256) r[d] = Seed[(long)t * kD + d];
    __syncthreads();
    const float* sb = slots + (((long)b * kB + k) * kM) * kD;
    for (int step = 0; step < 2; ++step){
      int g = tid >> 2, sub = tid & 3;
      const float* sp = sb + (long)g * kD + sub * 128;
      const float* rp = r + sub * 128;
      float p = 0.0f;
      for (int d = 0; d < 128; ++d) p += rp[d] * sp[d];
      red[tid] = p; __syncthreads();
      if (sub == 0) att[g] = (red[tid] + red[tid+1] + red[tid+2] + red[tid+3]) * scale;
      __syncthreads();
      if (tid == 0){
        float mx = -1e30f;
        for (int m = 0; m < kM; ++m) mx = fmaxf(mx, att[m]);
        float s = 0.0f;
        for (int m = 0; m < kM; ++m){ float e = __expf(att[m] - mx); att[m] = e; s += e; }
        float inv = 1.0f / s;
        for (int m = 0; m < kM; ++m) att[m] *= inv;
      }
      __syncthreads();
      for (int d = tid; d < kD; d += 256){
        float s = 0.0f;
        for (int m = 0; m < kM; ++m) s += att[m] * sb[(long)m * kD + d];
        r[d] = s;
      }
      __syncthreads();
    }
    for (int d = tid; d < kD; d += 256) accv[d] += r[d];
    __syncthreads();
  }
  for (int d = tid; d < kD; d += 256) H[(long)t * kD + d] += accv[d];
}

// ---------------- final LN -> bf16 ----------------
__global__ void k_final_ln(const float* __restrict__ X, const float* __restrict__ g,
                           const float* __restrict__ bta, bf16_t* __restrict__ out){
  __shared__ float red[256]; __shared__ float stat[2];
  int t = blockIdx.x; int tid = threadIdx.x;
  float x0 = X[(long)t * kD + tid];
  float x1 = X[(long)t * kD + tid + 256];
  red[tid] = x0 + x1; __syncthreads();
  for (int s = 128; s > 0; s >>= 1){ if (tid < s) red[tid] += red[tid + s]; __syncthreads(); }
  if (tid == 0) stat[0] = red[0] * (1.0f / kD);
  __syncthreads();
  float mu = stat[0];
  float d0 = x0 - mu, d1 = x1 - mu;
  red[tid] = d0 * d0 + d1 * d1; __syncthreads();
  for (int s = 128; s > 0; s >>= 1){ if (tid < s) red[tid] += red[tid + s]; __syncthreads(); }
  if (tid == 0) stat[1] = red[0] * (1.0f / kD);
  __syncthreads();
  float inv = rsqrtf(stat[1] + 1e-5f);
  out[(long)t * kD + tid]       = f2bf(d0 * inv * g[tid] + bta[tid]);
  out[(long)t * kD + tid + 256] = f2bf(d1 * inv * g[tid + 256] + bta[tid + 256]);
}

// =======================================================================
extern "C" void kernel_launch(void* const* d_in, const int* in_sizes, int n_in,
                              void* d_out, int out_size, void* d_ws, size_t ws_size,
                              hipStream_t stream){
  (void)in_sizes; (void)n_in; (void)out_size; (void)ws_size;
  const int*   ids      = (const int*)  d_in[0];
  const float* emb      = (const float*)d_in[1];
  const float* pos      = (const float*)d_in[2];
  const float* s1_Wv    = (const float*)d_in[3];
  const float* s1_Wg    = (const float*)d_in[4];
  const float* s1_alog  = (const float*)d_in[5];
  const float* s1_W1    = (const float*)d_in[6];
  const float* s1_W2    = (const float*)d_in[7];
  const float* s1_lng   = (const float*)d_in[8];
  const float* s1_lnb   = (const float*)d_in[9];
  const float* s3_Wv    = (const float*)d_in[10];
  const float* s3_Wg    = (const float*)d_in[11];
  const float* s3_alog  = (const float*)d_in[12];
  const float* s3_W1    = (const float*)d_in[13];
  const float* s3_W2    = (const float*)d_in[14];
  const float* s3_lng   = (const float*)d_in[15];
  const float* s3_lnb   = (const float*)d_in[16];
  const float* seedw_W  = (const float*)d_in[17];
  const float* seedw_b  = (const float*)d_in[18];
  const float* pcm_Wpred= (const float*)d_in[19];
  const float* pcm_bpred= (const float*)d_in[20];
  const float* pcm_Wgain= (const float*)d_in[21];
  const float* wnov_W   = (const float*)d_in[22];
  const float* wnov_b   = (const float*)d_in[23];
  const float* pm_Wwrite= (const float*)d_in[24];
  const float* pm_Wkey  = (const float*)d_in[25];
  const float* pm_state = (const float*)d_in[26];
  const float* em_slots = (const float*)d_in[27];
  const float* lnf_g    = (const float*)d_in[28];
  const float* lnf_b    = (const float*)d_in[29];

  float* logits = (float*)d_out;
  float* aux    = logits + (size_t)kNT * kV;

  char* wp = (char*)d_ws;
  auto take = [&](size_t bytes)->char*{ char* p = wp; wp += (bytes + 255) & ~(size_t)255; return p; };
  float*  X0    = (float*) take(sizeof(float) * (size_t)kNT * kD);
  float*  H     = (float*) take(sizeof(float) * (size_t)kNT * kD);
  float*  Vb    = (float*) take(sizeof(float) * (size_t)kNT * kD);
  float*  Gb    = (float*) take(sizeof(float) * (size_t)kNT * kD);
  float*  Ub    = (float*) take(sizeof(float) * (size_t)kNT * kD);
  float*  Surp  = (float*) take(sizeof(float) * (size_t)kNT * kD);
  float*  Seed  = (float*) take(sizeof(float) * (size_t)kNT * kD);
  float*  Wc    = (float*) take(sizeof(float) * (size_t)kNT * kD);
  float*  CumPM = (float*) take(sizeof(float) * (size_t)kNT * kB * kD);
  float*  CumEM = (float*) take(sizeof(float) * (size_t)kNT * kB * kD);
  float*  InvN  = (float*) take(sizeof(float) * (size_t)kBS * kB * kM);
  bf16_t* TanhS = (bf16_t*)take(sizeof(bf16_t) * (size_t)kNT * kD);
  bf16_t* WwrT  = (bf16_t*)take(sizeof(bf16_t) * (size_t)kB * kD * kD);
  bf16_t* AlnBf = (bf16_t*)take(sizeof(bf16_t) * (size_t)kNT * kD);
  bf16_t* EmbBf = (bf16_t*)take(sizeof(bf16_t) * (size_t)kV * kD);

  k_zero_aux<<<1, 1, 0, stream>>>(aux);
  k_embed<<<kNT, 256, 0, stream>>>(ids, emb, pos, X0, H);

  for (int l = 0; l < kL; ++l){
    k_scan_pre<<<kNT * kC, 64, 0, stream>>>(H, s1_Wv + (size_t)l*kC*kDC*kDC,
        s1_Wg + (size_t)l*kC*kDC*kDC, s1_lng + (size_t)l*kDC, s1_lnb + (size_t)l*kDC, Vb, Gb);
    k_scan_seq<<<kBS * kC, 64, 0, stream>>>(Vb, Gb, s1_alog + (size_t)l*kC*kDC, Ub);
    k_scan_mlp<<<kNT * kC, 128, 0, stream>>>(Ub, s1_W1 + (size_t)l*kC*kDC*kE,
        s1_W2 + (size_t)l*kC*kE*kDC, H);
  }

  k_pcm<<<kNT * kC, 64, 0, stream>>>(H, X0, pcm_Wpred, pcm_bpred, pcm_Wgain, Surp, aux);
  k_seedw<<<kNT * kC, 128, 0, stream>>>(H, seedw_W, seedw_b, Seed, Wc);

  k_tanh_bf<<<1024, 256, 0, stream>>>(Surp, TanhS, (long)kNT * kD);
  k_wwrT<<<2048, 256, 0, stream>>>(pm_Wwrite, WwrT);
  // delta_pm = ETA * tanh(surp) @ Wwrite[k]  -> CumPM (pre-cumsum), batched over 8 banks
  k_gemm_nt<<<dim3(kNT/32, kD/256, kB), 256, 0, stream>>>(TanhS, WwrT, CumPM,
      kD, kB * kD, (long)kD * kD, (long)kD, kETA);
  k_cumsum<<<kBS * kB, 512, 0, stream>>>(CumPM);

  k_slotnorm<<<kBS * kB * kM, 64, 0, stream>>>(em_slots, InvN);
  k_em_write<<<kNT, 256, 0, stream>>>(H, Wc, wnov_W, wnov_b, em_slots, InvN, CumEM);
  k_cumsum<<<kBS * kB, 512, 0, stream>>>(CumEM);

  k_pm_read<<<kNT, 256, 0, stream>>>(H, pm_Wkey, pm_state, CumPM, CumEM);
  k_em_read<<<kNT, 256, 0, stream>>>(Seed, em_slots, H);

  for (int l = 0; l < kL; ++l){
    k_scan_pre<<<kNT * kC, 64, 0, stream>>>(H, s3_Wv + (size_t)l*kC*kDC*kDC,
        s3_Wg + (size_t)l*kC*kDC*kDC, s3_lng + (size_t)l*kDC, s3_lnb + (size_t)l*kDC, Vb, Gb);
    k_scan_seq<<<kBS * kC, 64, 0, stream>>>(Vb, Gb, s3_alog + (size_t)l*kC*kDC, Ub);
    k_scan_mlp<<<kNT * kC, 128, 0, stream>>>(Ub, s3_W1 + (size_t)l*kC*kDC*kE,
        s3_W2 + (size_t)l*kC*kE*kDC, H);
  }

  k_final_ln<<<kNT, 256, 0, stream>>>(H, lnf_g, lnf_b, AlnBf);
  k_f32_bf<<<4096, 256, 0, stream>>>(emb, EmbBf, (long)kV * kD);
  // logits = LN(Hp) @ embedding^T  (tied embeddings), M=2048, N=32000, K=512
  k_gemm_nt<<<dim3(kNT/32, kV/256, 1), 256, 0, stream>>>(AlnBf, EmbBf, logits,
      kD, kV, 0, 0, 1.0f);
}
